// MACE_7739531067654
// MI455X (gfx1250) — matrix operand order repaired and checked by reading, not compiled, not run on hardware
//
#include <hip/hip_runtime.h>
#include <math.h>

// ---------------- problem constants ----------------
#define NN   10000
#define EE   160000
#define CC   64
#define SS   10
#define NBB  8
// scales
#define INV_SQRT_S   0.31622776601f       // 1/sqrt(10)
#define INV_SQRT_CS  0.03952847075f       // 1/sqrt(640)
#define INV_SQRT_C   0.125f               // 1/sqrt(64)
#define INV_SQRT_NB  0.35355339059f       // 1/sqrt(8)
#define AGG_LIN_SCL  0.0078125f           // (1/16 avg_nei) * (1/8)

typedef __attribute__((ext_vector_type(16))) _Float16 v16h;
typedef __attribute__((ext_vector_type(8)))  float    v8f;

// ---------------- WMMA helper (v_wmma_f32_16x16x32_f16) ----------------
__device__ __forceinline__ v8f wmma32(v16h a, v16h b, v8f c) {
  return __builtin_amdgcn_wmma_f32_16x16x32_f16(false, a, false, b, (short)0, c,
                                                false, false);
}

// ---- swizzled operand layouts: one contiguous v16h (32B) per lane ----
// B buffer dims: [KC][NT][32 lanes][16 halves]
__device__ __forceinline__ v16h load_B_swz(const _Float16* base, int kc, int nt,
                                           int NT, int lane) {
  return *(const v16h*)(base + (((kc * NT + nt) * 32) + lane) * 16);
}
// A buffer dims: [KC][32 lanes][16 halves]
__device__ __forceinline__ v16h load_A_swz(const _Float16* base, int kc, int lane) {
  return *(const v16h*)(base + ((kc * 32) + lane) * 16);
}

// position of logical A element (row, k) inside the A-swizzled buffer
__device__ __forceinline__ int a_swz_idx(int row, int k) {
  int kc = k >> 5, kk = k & 31;
  int hi = (kk >> 3) & 1;
  int e = (kk & 7) + (kk >= 16 ? 8 : 0);
  return ((kc * 32 + hi * 16 + row) << 4) + e;
}

// cooperative fill of a B-swizzled f16 weight block from row-major f32 (K=KC*32 x Nn)
__device__ __forceinline__ void fill_B_swz(const float* __restrict__ W, _Float16* dst,
                                           int KC, int NT, int tid, int nth) {
  int Nn = NT * 16;
  int total = KC * NT * 32 * 16;
  for (int i = tid; i < total; i += nth) {
    int e = i & 15;
    int lane = (i >> 4) & 31;
    int ntc = i >> 9;
    int nt = ntc % NT, kc = ntc / NT;
    int k = kc * 32 + (e < 8 ? e : e + 8) + 8 * (lane >> 4);
    int n = nt * 16 + (lane & 15);
    dst[i] = (_Float16)W[k * Nn + n];
  }
}

// fast silu: single v_exp_f32 + v_rcp_f32 (no IEEE-div expansion)
__device__ __forceinline__ float silu(float x) {
  return x * __builtin_amdgcn_rcpf(1.0f + __expf(-x));
}

// ---------------- kernel 1: species + embedding ----------------
__global__ void k_species_embed(const float* __restrict__ onehot,
                                const float* __restrict__ W_emb,
                                int* __restrict__ species,
                                float* __restrict__ feats) {
  int n = blockIdx.x * blockDim.x + threadIdx.x;
  if (n >= NN) return;
  int s = 0;
#pragma unroll
  for (int j = 1; j < SS; ++j)
    if (onehot[n * SS + j] > 0.5f) s = j;
  species[n] = s;
#pragma unroll 8
  for (int c = 0; c < CC; ++c)
    feats[n * CC + c] = W_emb[s * CC + c] * INV_SQRT_S;
}

// ---------------- kernel 2: edge featurization (sh, radial, keep) ----------------
__global__ void k_edge_feat(const float* __restrict__ ea,
                            const int* __restrict__ eidx,
                            float* __restrict__ sh,
                            float* __restrict__ ef,
                            float* __restrict__ keep) {
  int e = blockIdx.x * blockDim.x + threadIdx.x;
  if (e >= EE) return;
  float r = ea[e * 4 + 0];
  float x = ea[e * 4 + 1], y = ea[e * 4 + 2], z = ea[e * 4 + 3];
  float rn = sqrtf(x * x + y * y + z * z);
  float inv = __builtin_amdgcn_rcpf(fmaxf(rn, 1e-12f));
  x *= inv; y *= inv; z *= inv;

  const float s3 = 1.73205080757f, s5 = 2.23606797750f, s7 = 2.64575131106f;
  const float s15 = 3.87298334621f, s42o6 = 1.08012344973f, s168o8 = 1.62018517460f;
  float sh20 = s15 * x * z;
  float sh21 = s15 * x * y;
  float sh22 = s5 * (y * y - 0.5f * (x * x + z * z));
  float sh23 = s15 * y * z;
  float sh24 = 0.5f * s15 * (z * z - x * x);
  float o[16];
  o[0] = 1.0f;
  o[1] = s3 * x; o[2] = s3 * y; o[3] = s3 * z;
  o[4] = sh20; o[5] = sh21; o[6] = sh22; o[7] = sh23; o[8] = sh24;
  o[9]  = s42o6 * (sh20 * z + sh24 * x);
  o[10] = s7 * sh20 * y;
  o[11] = s168o8 * (4.0f * y * y - x * x - z * z) * x;
  o[12] = 0.5f * s7 * y * (2.0f * y * y - 3.0f * x * x - 3.0f * z * z);
  o[13] = s168o8 * z * (4.0f * y * y - x * x - z * z);
  o[14] = s7 * sh24 * y;
  o[15] = s42o6 * (sh24 * z - sh20 * x);
#pragma unroll
  for (int j = 0; j < 16; ++j) sh[e * 16 + j] = o[j];

  // radial: bessel * polynomial envelope (p=5, r_max=5)
  float rr = fmaxf(r, 1e-9f);
  float t = r * 0.2f;
  float t2 = t * t, t5 = t2 * t2 * t;
  float env = 1.0f - 21.0f * t5 + 35.0f * t5 * t - 15.0f * t5 * t2;
  if (r >= 5.0f) env = 0.0f;
  float pref = 0.63245553203f * __builtin_amdgcn_rcpf(rr);  // sqrt(2/5)/rr
#pragma unroll
  for (int k = 1; k <= NBB; ++k)
    ef[e * NBB + k - 1] = pref * __sinf((float)k * 0.62831853072f * rr) * env;

  int snd = eidx[e], rcv = eidx[EE + e];
  keep[e] = ((fabsf(r) < 1e-7f) && (snd == rcv)) ? 0.0f : 1.0f;
}

// ---------------- kernel 3: species-dependent skip (scalar GEMV) ----------------
__global__ void k_sc(const float* __restrict__ feats,
                     const int* __restrict__ species,
                     const float* __restrict__ W_skip,  // (C,S,C) for this iter
                     float* __restrict__ sc) {
  int idx = blockIdx.x * blockDim.x + threadIdx.x;  // n*64+m
  if (idx >= NN * CC) return;
  int n = idx >> 6, m = idx & 63;
  int s = species[n];
  float acc = 0.0f;
#pragma unroll 8
  for (int k = 0; k < CC; ++k)
    acc += feats[n * CC + k] * W_skip[(k * SS + s) * CC + m];
  sc[idx] = acc * INV_SQRT_CS;
}

// ---------------- kernel 4: up = feats @ W_up / sqrt(C)  (WMMA) ----------------
template <int COPY0>
__global__ __launch_bounds__(128) void k_node_up(const float* __restrict__ feats,
                                                 const float* __restrict__ W_up,
                                                 float* __restrict__ up,
                                                 float* __restrict__ up0_out) {
  __shared__ __align__(32) _Float16 sW[2 * 4 * 32 * 16];   // B-swizzled 64x64
  __shared__ __align__(32) _Float16 sF[4][2 * 32 * 16];    // A-swizzled 16x64
  int tid = threadIdx.x;
  fill_B_swz(W_up, sW, 2, 4, tid, blockDim.x);
  __syncthreads();

  int wave = tid >> 5, lane = tid & 31;
  int t = blockIdx.x * 4 + wave;            // 16-node tile, wave-uniform guard
  if (t >= NN / 16) return;
  _Float16* fb = sF[wave];
  for (int i = lane; i < 16 * CC; i += 32) {
    int row = i >> 6, k = i & 63;
    fb[a_swz_idx(row, k)] = (_Float16)feats[(t * 16 + row) * CC + k];
  }
  __builtin_amdgcn_wave_barrier();

  v16h alo = load_A_swz(fb, 0, lane);
  v16h ahi = load_A_swz(fb, 1, lane);
  int col = lane & 15, hi = lane >> 4;
#pragma unroll
  for (int nt = 0; nt < 4; ++nt) {
    v8f c = {};
    c = wmma32(alo, load_B_swz(sW, 0, nt, 4, lane), c);
    c = wmma32(ahi, load_B_swz(sW, 1, nt, 4, lane), c);
#pragma unroll
    for (int r = 0; r < 8; ++r) {
      int node = t * 16 + r + 8 * hi;
      float v = c[r] * INV_SQRT_C;
      up[node * CC + nt * 16 + col] = v;
      if (COPY0) up0_out[node * CC + nt * 16 + col] = v;
    }
  }
}

// ---------------- kernel 5: fused radial MLP (WMMA) + message scatter ----------------
__global__ __launch_bounds__(128) void k_radial_msg(const float* __restrict__ ef,
                                                    const float* __restrict__ sh,
                                                    const float* __restrict__ keep,
                                                    const float* __restrict__ up,
                                                    const int* __restrict__ eidx,
                                                    const float* __restrict__ M0,
                                                    const float* __restrict__ M1,
                                                    const float* __restrict__ M2,
                                                    const float* __restrict__ M3,
                                                    float* __restrict__ agg) {
  __shared__ __align__(32) _Float16 sM0[1 * 4 * 32 * 16];    // K padded 8->32, B-swz
  __shared__ __align__(32) _Float16 sM1[2 * 4 * 32 * 16];
  __shared__ __align__(32) _Float16 sM2[2 * 4 * 32 * 16];
  __shared__ __align__(32) _Float16 sM3[2 * 16 * 32 * 16];
  __shared__ __align__(32) _Float16 sH[4][2 * 32 * 16];      // per-wave A-swz tile

  int tid = threadIdx.x, nth = blockDim.x;
  // M0: pad K from 8 to 32 with zeros, B-swizzled
  for (int i = tid; i < 4 * 32 * 16; i += nth) {
    int e = i & 15;
    int lane2 = (i >> 4) & 31;
    int nt = i >> 9;
    int k = (e < 8 ? e : e + 8) + 8 * (lane2 >> 4);
    int n = nt * 16 + (lane2 & 15);
    sM0[i] = (k < NBB) ? (_Float16)M0[k * 64 + n] : (_Float16)0.0f;
  }
  fill_B_swz(M1, sM1, 2, 4, tid, nth);
  fill_B_swz(M2, sM2, 2, 4, tid, nth);
  fill_B_swz(M3, sM3, 2, 16, tid, nth);
  __syncthreads();

  int wave = tid >> 5, lane = tid & 31;
  int e0 = (blockIdx.x * 4 + wave) * 16;    // wave-uniform
  if (e0 >= EE) return;
  _Float16* hb = sH[wave];
  int row = lane & 15, hi = lane >> 4, col = lane & 15;

  // ---- layer 0: (16 x 8pad32) @ (32 x 64), silu ----
  v16h a0;
#pragma unroll
  for (int e = 0; e < 16; ++e) {
    int k = (e < 8 ? e : e + 8) + hi * 8;
    a0[e] = (k < NBB) ? (_Float16)ef[(e0 + row) * NBB + k] : (_Float16)0.0f;
  }
#pragma unroll
  for (int nt = 0; nt < 4; ++nt) {
    v8f c = {};
    c = wmma32(a0, load_B_swz(sM0, 0, nt, 4, lane), c);
#pragma unroll
    for (int r = 0; r < 8; ++r)
      hb[a_swz_idx(r + 8 * hi, nt * 16 + col)] = (_Float16)silu(c[r] * INV_SQRT_NB);
  }
  __builtin_amdgcn_wave_barrier();

  // ---- layers 1 & 2: 64 -> 64, silu ----
#pragma unroll 1
  for (int layer = 0; layer < 2; ++layer) {
    const _Float16* sM = (layer == 0) ? sM1 : sM2;   // runtime select
    v16h alo = load_A_swz(hb, 0, lane);
    v16h ahi = load_A_swz(hb, 1, lane);
    v8f acc[4];
#pragma unroll
    for (int nt = 0; nt < 4; ++nt) {
      v8f c = {};
      c = wmma32(alo, load_B_swz(sM, 0, nt, 4, lane), c);
      c = wmma32(ahi, load_B_swz(sM, 1, nt, 4, lane), c);
      acc[nt] = c;
    }
    __builtin_amdgcn_wave_barrier();
#pragma unroll
    for (int nt = 0; nt < 4; ++nt)
#pragma unroll
      for (int r = 0; r < 8; ++r)
        hb[a_swz_idx(r + 8 * hi, nt * 16 + col)] = (_Float16)silu(acc[nt][r] * INV_SQRT_C);
    __builtin_amdgcn_wave_barrier();
  }

  // ---- layer 3: 64 -> 256 + fused message scatter ----
  v16h alo = load_A_swz(hb, 0, lane);
  v16h ahi = load_A_swz(hb, 1, lane);

  int snd[8], rcv[8];
  float kp[8];
#pragma unroll
  for (int r = 0; r < 8; ++r) {
    int e = e0 + r + 8 * hi;
    snd[r] = eidx[e];
    rcv[r] = eidx[EE + e];
    kp[r] = keep[e];
  }

#pragma unroll 1
  for (int nt = 0; nt < 16; ++nt) {
    v8f c = {};
    c = wmma32(alo, load_B_swz(sM3, 0, nt, 16, lane), c);
    c = wmma32(ahi, load_B_swz(sM3, 1, nt, 16, lane), c);
    int j = nt * 16 + col;        // 0..255
    int ch = j >> 2, l = j & 3;   // channel, angular order
    int ms = (l == 0) ? 0 : (l == 1 ? 1 : (l == 2 ? 4 : 9));
    int me = (l == 0) ? 1 : (l == 1 ? 4 : (l == 2 ? 9 : 16));
#pragma unroll
    for (int r = 0; r < 8; ++r) {
      int e = e0 + r + 8 * hi;
      float wv = c[r] * INV_SQRT_C;
      float base = up[snd[r] * CC + ch] * wv * kp[r];
      float* aggp = agg + ((long)rcv[r] * CC + ch) * 16;
      for (int m = ms; m < me; ++m)
        atomicAdd(&aggp[m], base * sh[e * 16 + m]);
    }
  }
}

// ---------------- kernel 6: per-m channel mixing A = agg @ W_lin (WMMA, in-place) ----------------
__global__ __launch_bounds__(128) void k_lin(float* __restrict__ agg,
                                             const float* __restrict__ W_lin) {  // (4,C,C)
  __shared__ __align__(32) _Float16 sW[4][2 * 4 * 32 * 16];  // 4 B-swizzled 64x64
  __shared__ __align__(32) _Float16 sT[4][2 * 32 * 16];
  int tid = threadIdx.x;
  for (int l = 0; l < 4; ++l)
    fill_B_swz(W_lin + l * CC * CC, sW[l], 2, 4, tid, blockDim.x);
  __syncthreads();

  int wave = tid >> 5, lane = tid & 31;
  int task = blockIdx.x * 4 + wave;          // task = tile*16 + m (wave-uniform)
  if (task >= (NN / 16) * 16) return;
  int t = task >> 4, m = task & 15;
  int l = (m == 0) ? 0 : (m < 4 ? 1 : (m < 9 ? 2 : 3));
  _Float16* tb = sT[wave];

  for (int i = lane; i < 16 * CC; i += 32) {
    int nn = i >> 6, cc2 = i & 63;
    tb[a_swz_idx(nn, cc2)] = (_Float16)agg[((long)(t * 16 + nn) * CC + cc2) * 16 + m];
  }
  __builtin_amdgcn_wave_barrier();

  v16h alo = load_A_swz(tb, 0, lane);
  v16h ahi = load_A_swz(tb, 1, lane);
  int col = lane & 15, hi = lane >> 4;
#pragma unroll
  for (int nt = 0; nt < 4; ++nt) {
    v8f c = {};
    c = wmma32(alo, load_B_swz(sW[l], 0, nt, 4, lane), c);
    c = wmma32(ahi, load_B_swz(sW[l], 1, nt, 4, lane), c);
#pragma unroll
    for (int r = 0; r < 8; ++r)
      agg[((long)(t * 16 + r + 8 * hi) * CC + nt * 16 + col) * 16 + m] =
          c[r] * AGG_LIN_SCL;   // folds /AVG_NEI and /sqrt(C)
  }
}

// ---------------- kernel 7: invariants + b @ W_prod + sc (WMMA) ----------------
__global__ __launch_bounds__(128) void k_prod(const float* __restrict__ agg,  // holds A now
                                              const float* __restrict__ sc,
                                              const int* __restrict__ species,
                                              const float* __restrict__ P1,   // (S,C)
                                              const float* __restrict__ P2,   // (S,4,C)
                                              const float* __restrict__ P3,
                                              const float* __restrict__ W_prod,
                                              float* __restrict__ feats,
                                              float* __restrict__ out_i) {
  __shared__ __align__(32) _Float16 sW[2 * 4 * 32 * 16];
  __shared__ __align__(32) _Float16 sB[4][2 * 32 * 16];
  int tid = threadIdx.x;
  fill_B_swz(W_prod, sW, 2, 4, tid, blockDim.x);
  __syncthreads();

  int wave = tid >> 5, lane = tid & 31;
  int t = blockIdx.x * 4 + wave;
  if (t >= NN / 16) return;
  _Float16* bb = sB[wave];

  // compute b for this 16-node tile (32 (n,d) pairs per lane)
  for (int q = 0; q < 32; ++q) {
    int idx = q * 32 + lane;                 // 0..1023
    int nn = idx >> 6, d = idx & 63;
    long node = (long)t * 16 + nn;
    const float* Ap = agg + (node * CC + d) * 16;
    float A0 = Ap[0];
    float iv[4];
    iv[0] = A0 * A0;
    iv[1] = Ap[1] * Ap[1] + Ap[2] * Ap[2] + Ap[3] * Ap[3];
    iv[2] = Ap[4] * Ap[4] + Ap[5] * Ap[5] + Ap[6] * Ap[6] + Ap[7] * Ap[7] + Ap[8] * Ap[8];
    iv[3] = 0.0f;
#pragma unroll
    for (int m = 9; m < 16; ++m) iv[3] += Ap[m] * Ap[m];
    int s = species[node];
    float w1 = P1[s * CC + d];
    float s2 = 0.0f, s3v = 0.0f;
#pragma unroll
    for (int l = 0; l < 4; ++l) {
      s2 += iv[l] * P2[(s * 4 + l) * CC + d];
      s3v += iv[l] * P3[(s * 4 + l) * CC + d];
    }
    bb[a_swz_idx(nn, d)] = (_Float16)(A0 * w1 + s2 + A0 * s3v);
  }
  __builtin_amdgcn_wave_barrier();

  v16h alo = load_A_swz(bb, 0, lane);
  v16h ahi = load_A_swz(bb, 1, lane);
  int col = lane & 15, hi = lane >> 4;
#pragma unroll
  for (int nt = 0; nt < 4; ++nt) {
    v8f c = {};
    c = wmma32(alo, load_B_swz(sW, 0, nt, 4, lane), c);
    c = wmma32(ahi, load_B_swz(sW, 1, nt, 4, lane), c);
#pragma unroll
    for (int r = 0; r < 8; ++r) {
      int node = t * 16 + r + 8 * hi;
      float v = c[r] * INV_SQRT_C + sc[node * CC + nt * 16 + col];
      feats[node * CC + nt * 16 + col] = v;
      out_i[node * CC + nt * 16 + col] = v;
    }
  }
}

// ---------------- launcher ----------------
extern "C" void kernel_launch(void* const* d_in, const int* in_sizes, int n_in,
                              void* d_out, int out_size, void* d_ws, size_t ws_size,
                              hipStream_t stream) {
  const float* edge_attr = (const float*)d_in[0];
  const float* onehot    = (const float*)d_in[1];
  const float* W_emb     = (const float*)d_in[2];
  const float* W_skip    = (const float*)d_in[3];
  const float* W_up      = (const float*)d_in[4];
  const float* M0        = (const float*)d_in[5];
  const float* M1        = (const float*)d_in[6];
  const float* M2        = (const float*)d_in[7];
  const float* M3        = (const float*)d_in[8];
  const float* W_lin     = (const float*)d_in[9];
  const float* P1        = (const float*)d_in[10];
  const float* P2        = (const float*)d_in[11];
  const float* P3        = (const float*)d_in[12];
  const float* W_prod    = (const float*)d_in[13];
  const int*   eidx      = (const int*)d_in[14];
  float* out = (float*)d_out;

  char* ws = (char*)d_ws;
  size_t off = 0;
  auto take = [&](size_t nbytes) {
    char* p = ws + off;
    off = (off + nbytes + 255) & ~(size_t)255;
    return p;
  };
  int*   species = (int*)take((size_t)NN * 4);
  float* feats   = (float*)take((size_t)NN * CC * 4);
  float* sc      = (float*)take((size_t)NN * CC * 4);
  float* up      = (float*)take((size_t)NN * CC * 4);
  float* sh      = (float*)take((size_t)EE * 16 * 4);
  float* ef      = (float*)take((size_t)EE * NBB * 4);
  float* keep    = (float*)take((size_t)EE * 4);
  float* agg     = (float*)take((size_t)NN * CC * 16 * 4);

  k_species_embed<<<(NN + 255) / 256, 256, 0, stream>>>(onehot, W_emb, species, feats);
  k_edge_feat<<<EE / 256, 256, 0, stream>>>(edge_attr, eidx, sh, ef, keep);

  for (int i = 0; i < 2; ++i) {
    k_sc<<<(NN * CC) / 256, 256, 0, stream>>>(feats, species, W_skip + (size_t)i * CC * SS * CC, sc);
    if (i == 0)
      k_node_up<1><<<(NN / 16 + 3) / 4, 128, 0, stream>>>(feats, W_up, up,
                                                          out + 2 * (size_t)NN * CC);
    else
      k_node_up<0><<<(NN / 16 + 3) / 4, 128, 0, stream>>>(feats, W_up + (size_t)CC * CC,
                                                          up, nullptr);
    (void)hipMemsetAsync(agg, 0, (size_t)NN * CC * 16 * 4, stream);
    k_radial_msg<<<EE / 64, 128, 0, stream>>>(ef, sh, keep, up, eidx,
                                              M0 + (size_t)i * NBB * 64,
                                              M1 + (size_t)i * 64 * 64,
                                              M2 + (size_t)i * 64 * 64,
                                              M3 + (size_t)i * 64 * 256, agg);
    k_lin<<<((NN / 16) * 16) / 4, 128, 0, stream>>>(agg, W_lin + (size_t)i * 4 * CC * CC);
    k_prod<<<(NN / 16 + 3) / 4, 128, 0, stream>>>(agg, sc, species,
                                                  P1 + (size_t)i * SS * CC,
                                                  P2 + (size_t)i * SS * 4 * CC,
                                                  P3 + (size_t)i * SS * 4 * CC,
                                                  W_prod + (size_t)i * CC * CC,
                                                  feats, out + (size_t)i * NN * CC);
  }
}
